// Attention_36215164239929
// MI455X (gfx1250) — matrix operand order repaired and checked by reading, not compiled, not run on hardware
//
#include <hip/hip_runtime.h>

// ---------------------------------------------------------------------------
// Problem constants (from reference): B=2, T=2048, C=1024, H=16, D=64
// ---------------------------------------------------------------------------
static constexpr int BB = 2;
static constexpr int TT = 2048;
static constexpr int CC = 1024;
static constexpr int HH = 16;
static constexpr int DD = 64;      // head dim
static constexpr int BT = BB * TT; // 4096 rows

typedef __bf16 bf16;
typedef __attribute__((ext_vector_type(16))) __bf16 v16bf;
typedef __attribute__((ext_vector_type(4)))  __bf16 v4bf;
typedef __attribute__((ext_vector_type(8)))  float  v8f;
typedef __attribute__((ext_vector_type(4)))  float  v4f;
typedef __attribute__((ext_vector_type(4)))  int    v4i;

// 16 bf16 = 32 bytes = two 16B chunks
union Frag16 { v16bf v; v4f q[2]; };

__device__ __forceinline__ v8f wmma_bf16(v16bf a, v16bf b, v8f c) {
    return __builtin_amdgcn_wmma_f32_16x16x32_bf16(false, a, false, b,
                                                   (short)0, c, false, false);
}

// ---- CDNA5 async global->LDS copy path (guarded; falls back to sync copy) --
#if defined(__AMDGCN__) && __has_builtin(__builtin_amdgcn_global_load_async_to_lds_b128)
#define USE_ASYNC_LDS 1
#else
#define USE_ASYNC_LDS 0
#endif

typedef __attribute__((address_space(1))) v4i as1_v4i;  // global int4*
typedef __attribute__((address_space(3))) v4i as3_v4i;  // LDS int4*

__device__ __forceinline__ void wait_async0() {
#if USE_ASYNC_LDS
#if __has_builtin(__builtin_amdgcn_s_wait_asynccnt)
    __builtin_amdgcn_s_wait_asynccnt(0);
#else
    asm volatile("s_wait_asynccnt 0x0" ::: "memory");
#endif
#endif
}

__device__ __forceinline__ void copy16_g2l(const bf16* g, bf16* l) {
#if USE_ASYNC_LDS
    __builtin_amdgcn_global_load_async_to_lds_b128(
        (as1_v4i*)(void*)g, (as3_v4i*)(void*)l, 0, 0);
#else
    *(v4f*)l = *(const v4f*)g;
#endif
}

// ---------------------------------------------------------------------------
// Kernel: elementwise f32 -> bf16 (vectorized x4)
// ---------------------------------------------------------------------------
__global__ __launch_bounds__(256)
void f32_to_bf16_kernel(const float* __restrict__ src, bf16* __restrict__ dst)
{
    size_t idx = ((size_t)blockIdx.x * 256 + threadIdx.x) * 4;
    v4f f = *(const v4f*)(src + idx);
    v4bf o = { (bf16)f.x, (bf16)f.y, (bf16)f.z, (bf16)f.w };
    *(v4bf*)(dst + idx) = o;
}

// ---------------------------------------------------------------------------
// Kernel: transpose f32 [K][N] -> bf16 [N][K] via LDS (coalesced both ways)
// ---------------------------------------------------------------------------
__global__ __launch_bounds__(256)
void transpose_f32_to_bf16_kernel(const float* __restrict__ src,
                                  bf16* __restrict__ dst,
                                  int K, int N)
{
    __shared__ float tile[32][33];
    const int k0 = blockIdx.y * 32;
    const int n0 = blockIdx.x * 32;
    const int tx = threadIdx.x & 31;
    const int ty = threadIdx.x >> 5;    // 0..7
    #pragma unroll
    for (int i = 0; i < 32; i += 8)
        tile[ty + i][tx] = src[(size_t)(k0 + ty + i) * N + n0 + tx];
    __syncthreads();
    #pragma unroll
    for (int i = 0; i < 32; i += 8)
        dst[(size_t)(n0 + ty + i) * K + k0 + tx] = (bf16)tile[tx][ty + i];
}

// ---------------------------------------------------------------------------
// Kernel: GEMM Out[M,N](f32) = A[M,K](bf16) @ WT[N,K](bf16, pre-transposed).
// 256 threads = 8 waves. Block tile 128x128, wave tile 32x64 (2x4 WMMA),
// K-step 32, double-buffered LDS staged with async global->LDS b128 copies.
// ---------------------------------------------------------------------------
__global__ __launch_bounds__(256)
void gemm_bf16_async_kernel(const bf16* __restrict__ A,
                            const bf16* __restrict__ WT,
                            float* __restrict__ Out,
                            int M, int N, int K)
{
    constexpr int BK = 32;
    __shared__ __align__(16) bf16 As[2][128 * BK];
    __shared__ __align__(16) bf16 Bs[2][128 * BK];

    const int tid   = threadIdx.x;
    const int lane  = tid & 31;
    const int wid   = tid >> 5;
    const int waveM = wid & 3;           // 4 x 32 rows
    const int waveN = wid >> 2;          // 2 x 64 cols
    const int halfw = (lane >> 4) & 1;
    const int l15   = lane & 15;

    const int blockM = blockIdx.y * 128;
    const int blockN = blockIdx.x * 128;
    (void)M;

    v8f acc[2][4] = {};

    auto stage = [&](int buf, int k0) {
        #pragma unroll
        for (int i = 0; i < 2; ++i) {
            int chunk = tid + i * 256;       // 0..511
            int row   = chunk >> 2;          // 0..127
            int k8    = (chunk & 3) << 3;    // 0,8,16,24
            copy16_g2l(A  + (size_t)(blockM + row) * K + k0 + k8,
                       &As[buf][row * BK + k8]);
            copy16_g2l(WT + (size_t)(blockN + row) * K + k0 + k8,
                       &Bs[buf][row * BK + k8]);
        }
    };

    const int nk = K / BK;
    stage(0, 0);
    wait_async0();
    __syncthreads();

    for (int kt = 0; kt < nk; ++kt) {
        const int cur = kt & 1;
        if (kt + 1 < nk) stage(cur ^ 1, (kt + 1) * BK);

        Frag16 af[2], bfr[4];
        #pragma unroll
        for (int mt = 0; mt < 2; ++mt) {
            // A layout: half 0 -> K {0..7,16..23}; half 1 -> {8..15,24..31}
            const bf16* p = &As[cur][(waveM * 32 + mt * 16 + l15) * BK + (halfw ? 8 : 0)];
            af[mt].q[0] = *(const v4f*)p;
            af[mt].q[1] = *(const v4f*)(p + 16);
        }
        #pragma unroll
        for (int nt = 0; nt < 4; ++nt) {
            // B layout: half 0 -> K {0..7,16..23}; half 1 -> {8..15,24..31}
            const bf16* p = &Bs[cur][(waveN * 64 + nt * 16 + l15) * BK + (halfw ? 8 : 0)];
            bfr[nt].q[0] = *(const v4f*)p;
            bfr[nt].q[1] = *(const v4f*)(p + 16);
        }
        #pragma unroll
        for (int mt = 0; mt < 2; ++mt)
            #pragma unroll
            for (int nt = 0; nt < 4; ++nt)
                acc[mt][nt] = wmma_bf16(af[mt].v, bfr[nt].v, acc[mt][nt]);

        wait_async0();
        __syncthreads();
    }

    #pragma unroll
    for (int mt = 0; mt < 2; ++mt) {
        #pragma unroll
        for (int nt = 0; nt < 4; ++nt) {
            int rbase = blockM + waveM * 32 + mt * 16 + halfw * 8;
            int col   = blockN + waveN * 64 + nt * 16 + l15;
            #pragma unroll
            for (int i = 0; i < 8; ++i)
                Out[(size_t)(rbase + i) * N + col] = acc[mt][nt][i];
        }
    }
}

// ---------------------------------------------------------------------------
// Kernel: split qkv[BT,3C](f32) -> bf16 q[bh][t][d], k[bh][t][d], vT[bh][d][t]
// One block per (bh, 64-row t-chunk); V transposed through LDS (coalesced).
// ---------------------------------------------------------------------------
__global__ __launch_bounds__(256)
void qkv_split_kernel(const float* __restrict__ qkv,
                      bf16* __restrict__ qb,
                      bf16* __restrict__ kb,
                      bf16* __restrict__ vtb)
{
    __shared__ float vt[64][65];
    const int blk = blockIdx.x;                 // 0 .. B*H*T/64-1
    const int bh  = blk / (TT / 64);
    const int t0  = (blk % (TT / 64)) * 64;
    const int b   = bh / HH, h = bh % HH;
    const int tx  = threadIdx.x & 63;
    const int ty  = threadIdx.x >> 6;           // 0..3

    #pragma unroll
    for (int i = 0; i < 64; i += 4) {
        int tl = ty + i;
        int t  = t0 + tl;
        size_t base = ((size_t)(b * TT + t)) * (3 * CC) + h * DD;
        float q = qkv[base + tx];
        float k = qkv[base + CC + tx];
        float v = qkv[base + 2 * CC + tx];
        qb[((size_t)bh * TT + t) * DD + tx] = (bf16)q;
        kb[((size_t)bh * TT + t) * DD + tx] = (bf16)k;
        vt[tl][tx] = v;
    }
    __syncthreads();
    #pragma unroll
    for (int i = 0; i < 64; i += 4) {
        int d = ty + i;
        vtb[((size_t)bh * DD + d) * TT + t0 + tx] = (bf16)vt[tx][d];
    }
}

// ---------------------------------------------------------------------------
// Kernel: flash attention. 128 threads = 4 waves; one wave per 16-query tile.
// 32-key blocks: QK^T (4 WMMA) -> online softmax -> PV (4 WMMA).
// Writes bf16 y[bt][h*D+d] for the output projection GEMM.
// ---------------------------------------------------------------------------
__global__ __launch_bounds__(128)
void flash_attn_kernel(const bf16* __restrict__ qb,
                       const bf16* __restrict__ kbuf,
                       const bf16* __restrict__ vtb,
                       bf16* __restrict__ y)
{
    __shared__ __align__(16) bf16 Pbuf[4][16 * 32];   // per-wave P re-layout

    const int tid   = threadIdx.x;
    const int lane  = tid & 31;
    const int wid   = tid >> 5;           // 0..3
    const int halfw = (lane >> 4) & 1;
    const int l15   = lane & 15;

    const int qblk = blockIdx.x;                  // 0 .. B*H*T/64-1
    const int bh   = qblk / (TT / 64);
    const int q0   = (qblk % (TT / 64)) * 64 + wid * 16;

    const bf16* Q  = qb   + (size_t)bh * TT * DD;
    const bf16* Kp = kbuf + (size_t)bh * TT * DD;
    const bf16* Vt = vtb  + (size_t)bh * DD * TT;

    // Q A-fragments for both 32-wide d-chunks (kept in registers)
    Frag16 qf[2];
    #pragma unroll
    for (int dc = 0; dc < 2; ++dc) {
        int row = q0 + l15;
        int db  = dc * 32 + (halfw ? 8 : 0);
        qf[dc].q[0] = *(const v4f*)(Q + (size_t)row * DD + db);
        qf[dc].q[1] = *(const v4f*)(Q + (size_t)row * DD + db + 16);
    }

    float m[8], l[8];
    v8f o[4] = {};                     // 16x64 output tile
    #pragma unroll
    for (int i = 0; i < 8; ++i) { m[i] = -1e30f; l[i] = 0.0f; }

    bf16* P = Pbuf[wid];
    const float scale = 0.125f;        // 1/sqrt(64)

    for (int kb0 = 0; kb0 < TT; kb0 += 32) {
        // ---- S(16x32) = Q @ K^T ----
        v8f s[2] = {};
        #pragma unroll
        for (int nt = 0; nt < 2; ++nt) {
            #pragma unroll
            for (int dc = 0; dc < 2; ++dc) {
                Frag16 kf;
                int key = kb0 + nt * 16 + l15;
                int db  = dc * 32 + (halfw ? 8 : 0);
                kf.q[0] = *(const v4f*)(Kp + (size_t)key * DD + db);
                kf.q[1] = *(const v4f*)(Kp + (size_t)key * DD + db + 16);
                s[nt] = wmma_bf16(qf[dc].v, kf.v, s[nt]);
            }
        }
        #pragma unroll
        for (int nt = 0; nt < 2; ++nt)
            #pragma unroll
            for (int i = 0; i < 8; ++i) s[nt][i] *= scale;

        // ---- row max across 16 lanes holding each row ----
        float rmax[8];
        #pragma unroll
        for (int i = 0; i < 8; ++i) rmax[i] = fmaxf(s[0][i], s[1][i]);
        #pragma unroll
        for (int mask = 1; mask < 16; mask <<= 1)
            #pragma unroll
            for (int i = 0; i < 8; ++i)
                rmax[i] = fmaxf(rmax[i], __shfl_xor(rmax[i], mask, 32));

        float sf[8];
        #pragma unroll
        for (int i = 0; i < 8; ++i) {
            float mn = fmaxf(m[i], rmax[i]);
            sf[i] = __expf(m[i] - mn);
            m[i] = mn;
        }
        #pragma unroll
        for (int nt = 0; nt < 2; ++nt)
            #pragma unroll
            for (int i = 0; i < 8; ++i)
                s[nt][i] = __expf(s[nt][i] - m[i]);
        float rsum[8];
        #pragma unroll
        for (int i = 0; i < 8; ++i) rsum[i] = s[0][i] + s[1][i];
        #pragma unroll
        for (int mask = 1; mask < 16; mask <<= 1)
            #pragma unroll
            for (int i = 0; i < 8; ++i)
                rsum[i] += __shfl_xor(rsum[i], mask, 32);
        #pragma unroll
        for (int i = 0; i < 8; ++i) l[i] = l[i] * sf[i] + rsum[i];

        // ---- write P (C layout) to LDS row-major [m][key] as bf16 ----
        #pragma unroll
        for (int nt = 0; nt < 2; ++nt)
            #pragma unroll
            for (int i = 0; i < 8; ++i)
                P[(i + halfw * 8) * 32 + nt * 16 + l15] = (bf16)s[nt][i];

        // ---- rescale O ----
        #pragma unroll
        for (int dt = 0; dt < 4; ++dt)
            #pragma unroll
            for (int i = 0; i < 8; ++i) o[dt][i] *= sf[i];

        // ---- P A-fragment from LDS (same wave; DS ops in-order) ----
        Frag16 pf;
        {
            int kbb = halfw ? 8 : 0;
            pf.q[0] = *(const v4f*)(&P[l15 * 32 + kbb]);
            pf.q[1] = *(const v4f*)(&P[l15 * 32 + kbb + 16]);
        }
        // ---- O += P @ V ----
        #pragma unroll
        for (int dt = 0; dt < 4; ++dt) {
            Frag16 vf;
            int d    = dt * 16 + l15;
            int key0 = kb0 + (halfw ? 8 : 0);
            vf.q[0] = *(const v4f*)(Vt + (size_t)d * TT + key0);
            vf.q[1] = *(const v4f*)(Vt + (size_t)d * TT + key0 + 16);
            o[dt] = wmma_bf16(pf.v, vf.v, o[dt]);
        }
    }

    // ---- epilogue: y[b*T + q][h*D + d] = (bf16)(O / l) ----
    const int b = bh / HH, h = bh % HH;
    #pragma unroll
    for (int dt = 0; dt < 4; ++dt) {
        #pragma unroll
        for (int i = 0; i < 8; ++i) {
            float val = o[dt][i] / l[i];
            y[((size_t)b * TT + q0 + i + halfw * 8) * CC + h * DD + dt * 16 + l15]
                = (bf16)val;
        }
    }
}

// ---------------------------------------------------------------------------
// Host-side launch
// ---------------------------------------------------------------------------
extern "C" void kernel_launch(void* const* d_in, const int* in_sizes, int n_in,
                              void* d_out, int out_size, void* d_ws, size_t ws_size,
                              hipStream_t stream) {
    (void)in_sizes; (void)n_in; (void)out_size; (void)ws_size;
    const float* x      = (const float*)d_in[0];   // [B,T,C]
    const float* w_attn = (const float*)d_in[1];   // [C,3C]
    const float* w_proj = (const float*)d_in[2];   // [C,C]
    float* out = (float*)d_out;                    // [B,T,C]

    char* ws = (char*)d_ws;
    size_t off = 0;
    auto carve = [&](size_t bytes) -> void* {
        void* p = ws + off;
        off += (bytes + 255) & ~(size_t)255;
        return p;
    };
    bf16*  xb   = (bf16*) carve((size_t)BT * CC * sizeof(bf16));          // 8.4 MB
    bf16*  watT = (bf16*) carve((size_t)3 * CC * CC * sizeof(bf16));      // 6.3 MB
    bf16*  wpT  = (bf16*) carve((size_t)CC * CC * sizeof(bf16));          // 2.1 MB
    float* qkv  = (float*)carve((size_t)BT * 3 * CC * sizeof(float));     // 50.3 MB
    bf16*  qb   = (bf16*) carve((size_t)BT * CC * sizeof(bf16));          // 8.4 MB
    bf16*  kb   = (bf16*) carve((size_t)BT * CC * sizeof(bf16));          // 8.4 MB
    bf16*  vtb  = (bf16*) carve((size_t)BT * CC * sizeof(bf16));          // 8.4 MB
    bf16*  ybuf = (bf16*) carve((size_t)BT * CC * sizeof(bf16));          // 8.4 MB

    // 0) one-time operand conversion / weight transposition (amortized K times)
    f32_to_bf16_kernel<<<(BT * CC / 4) / 256, 256, 0, stream>>>(x, xb);
    {
        dim3 g((3 * CC) / 32, CC / 32);
        transpose_f32_to_bf16_kernel<<<g, 256, 0, stream>>>(w_attn, watT, CC, 3 * CC);
    }
    {
        dim3 g(CC / 32, CC / 32);
        transpose_f32_to_bf16_kernel<<<g, 256, 0, stream>>>(w_proj, wpT, CC, CC);
    }

    // 1) qkv = x @ w_attn    (M=4096, N=3072, K=1024)
    {
        dim3 grid((3 * CC) / 128, BT / 128);
        gemm_bf16_async_kernel<<<grid, 256, 0, stream>>>(xb, watT, qkv,
                                                         BT, 3 * CC, CC);
    }
    // 2) split + bf16 convert (+ V transpose via LDS)
    qkv_split_kernel<<<(BB * HH * TT) / 64, 256, 0, stream>>>(qkv, qb, kb, vtb);

    // 3) flash attention -> ybuf (bf16, merged heads)
    flash_attn_kernel<<<(BB * HH * TT) / 64, 128, 0, stream>>>(qb, kb, vtb, ybuf);

    // 4) out = ybuf @ w_proj  (M=4096, N=1024, K=1024)
    {
        dim3 grid(CC / 128, BT / 128);
        gemm_bf16_async_kernel<<<grid, 256, 0, stream>>>(ybuf, wpT, out,
                                                         BT, CC, CC);
    }
}
